// MICLoss_57647051047499
// MI455X (gfx1250) — matrix operand order repaired and checked by reading, not compiled, not run on hardware
//
#include <hip/hip_runtime.h>
#include <hip/hip_bf16.h>

#define BB 4
#define II 32
#define NN 4096
#define DD 64
#define NCHUNK 4
#define CHUNK_COLS (NN / NCHUNK)   // 1024 columns per wave

typedef __attribute__((ext_vector_type(16))) _Float16 v16h;
typedef __attribute__((ext_vector_type(8)))  _Float16 v8h;
typedef __attribute__((ext_vector_type(8)))  float    v8f;

// ---------------------------------------------------------------------------
// Kernel 1: convert x -> f16 copy, derive per-point instance id from label
// ---------------------------------------------------------------------------
__global__ __launch_bounds__(256) void mic_prep(const float* __restrict__ x,
                                                const unsigned char* __restrict__ label,
                                                _Float16* __restrict__ xh,
                                                int* __restrict__ inst) {
    int idx = blockIdx.x * blockDim.x + threadIdx.x;   // over B*N points
    if (idx >= BB * NN) return;
    int b = idx / NN, n = idx - b * NN;
    const float* src = x + (size_t)idx * DD;
    _Float16* dst = xh + (size_t)idx * DD;
#pragma unroll 8
    for (int k = 0; k < DD; ++k) dst[k] = (_Float16)src[k];
    int id = -1;
    for (int i = 0; i < II; ++i) {
        if (label[((size_t)b * II + i) * NN + n]) { id = i; break; }
    }
    inst[idx] = id;
}

// ---------------------------------------------------------------------------
// Kernel 2: fused sim-tile WMMA + per-row reductions.
// One wave: 16 rows x CHUNK_COLS columns. Never materializes sim.
//   partial[chunk][b*N+p] for: sq_in, e_tot, e_in
// ---------------------------------------------------------------------------
__global__ __launch_bounds__(256) void mic_main(const _Float16* __restrict__ xh,
                                                const int* __restrict__ inst,
                                                float* __restrict__ psq,
                                                float* __restrict__ ptot,
                                                float* __restrict__ pein) {
    const int lane = threadIdx.x & 31;
    const int widx = threadIdx.x >> 5;
    const int w = blockIdx.x * 8 + widx;        // global wave id, 0..4095
    const int chunk   = w & (NCHUNK - 1);
    const int rowtile = (w >> 2) & (NN / 16 - 1);
    const int b       = w >> 10;                // / (NCHUNK * N/16)
    const int hi = lane >> 4;                   // lane half: 0 or 1
    const int lo = lane & 15;

    const _Float16* xb = xh + (size_t)b * NN * DD;
    const int p0 = rowtile * 16;

    // A operand (16x64 f16, rows p0..p0+15). Layout: lane -> M = lane&15;
    // lanes 0-15 hold K=0..15(+32*kb), lanes 16-31 hold K=16..31(+32*kb).
    const _Float16* arow = xb + (size_t)(p0 + lo) * DD;
    v16h a0, a1;
    {
        a0 = *(const v16h*)(arow +      16 * hi);
        a1 = *(const v16h*)(arow + 32 + 16 * hi);
    }

    // Instance id for the 8 rows this lane's C registers cover (M = r + 8*hi).
    int rinst[8];
#pragma unroll
    for (int r = 0; r < 8; ++r) rinst[r] = inst[b * NN + p0 + r + 8 * hi];

    v8f accsq = {}, acctot = {}, accein = {};
    const int c0 = chunk * CHUNK_COLS;

    for (int nt = 0; nt < CHUNK_COLS / 16; ++nt) {
        const int ncol = c0 + nt * 16 + lo;     // column for this lane (both halves)
        const _Float16* brow = xb + (size_t)ncol * DD;
        // B operand (32x16, K-major per lane): lanes 0-15 K=0..15, 16-31 K=16..31.
        v16h bt0 = *(const v16h*)(brow +      16 * hi);   // kb=0
        v16h bt1 = *(const v16h*)(brow + 32 + 16 * hi);   // kb=1
        v8f c = {};
        c = __builtin_amdgcn_wmma_f32_16x16x32_f16(false, a0, false, bt0,
                                                   (short)0, c, false, false);
        c = __builtin_amdgcn_wmma_f32_16x16x32_f16(false, a1, false, bt1,
                                                   (short)0, c, false, false);
        const int cinst = inst[b * NN + ncol];
#pragma unroll
        for (int r = 0; r < 8; ++r) {
            float d = c[r];                     // sim[p0 + r + 8*hi][ncol]
            float e = __expf(d);                // TEM == 1.0
            acctot[r] += e;
            if (cinst >= 0 && cinst == rinst[r]) {
                accein[r] += e;
                float dm = d - 1.0f;
                accsq[r] += dm * dm;
            }
        }
    }

    // Per-row totals: reduce across the 16 lanes of each half.
#pragma unroll
    for (int r = 0; r < 8; ++r) {
        float s = accsq[r], t = acctot[r], e = accein[r];
#pragma unroll
        for (int m = 1; m < 16; m <<= 1) {
            s += __shfl_xor(s, m, 32);
            t += __shfl_xor(t, m, 32);
            e += __shfl_xor(e, m, 32);
        }
        if (lo == 0) {
            int p = p0 + r + 8 * hi;
            size_t k = (size_t)chunk * (BB * NN) + (size_t)b * NN + p;
            psq[k] = s; ptot[k] = t; pein[k] = e;
        }
    }
}

// ---------------------------------------------------------------------------
// Kernel 3: deterministic final reduction. One thread per (b, i) bin.
// ---------------------------------------------------------------------------
__global__ __launch_bounds__(128) void mic_final(const int* __restrict__ inst,
                                                 const float* __restrict__ psq,
                                                 const float* __restrict__ ptot,
                                                 const float* __restrict__ pein,
                                                 float* __restrict__ out) {
    __shared__ float partial[128];
    const int t = threadIdx.x;          // 0..127
    const int b = t >> 5, i = t & 31;
    float cnt = 0.f, ssq = 0.f, sln = 0.f;
    for (int n = 0; n < NN; ++n) {
        int p = b * NN + n;
        if (inst[p] == i) {
            cnt += 1.f;
            float sq = 0.f, tot = 0.f, ein = 0.f;
#pragma unroll
            for (int cch = 0; cch < NCHUNK; ++cch) {
                size_t k = (size_t)cch * (BB * NN) + p;
                sq += psq[k]; tot += ptot[k]; ein += pein[k];
            }
            ssq += sq;
            sln += __logf(fmaxf(tot - ein, 1e-30f));
        }
    }
    float valid = (cnt >= 5.f) ? 1.f : 0.f;
    float lp = ssq / fmaxf(cnt * cnt, 1.f);
    float ln = sln / fmaxf(cnt, 1.f);
    partial[t] = valid * (0.5f * lp + 0.5f * ln) / (float)(BB * II);
    __syncthreads();
    for (int s = 64; s > 0; s >>= 1) {
        if (t < s) partial[t] += partial[t + s];
        __syncthreads();
    }
    if (t == 0) out[0] = partial[0];
}

// ---------------------------------------------------------------------------
extern "C" void kernel_launch(void* const* d_in, const int* in_sizes, int n_in,
                              void* d_out, int out_size, void* d_ws, size_t ws_size,
                              hipStream_t stream) {
    const float*         x     = (const float*)d_in[0];          // [B,N,D] f32
    const unsigned char* label = (const unsigned char*)d_in[1];  // [B,I,N] bool (1B)
    float* out = (float*)d_out;

    char* ws = (char*)d_ws;
    const size_t XH_BYTES   = (size_t)BB * NN * DD * sizeof(_Float16); // 2 MB
    const size_t INST_BYTES = (size_t)BB * NN * sizeof(int);           // 64 KB
    const size_t PART_ELEMS = (size_t)NCHUNK * BB * NN;

    _Float16* xh   = (_Float16*)ws;
    int*      inst = (int*)(ws + XH_BYTES);
    float*    psq  = (float*)(ws + XH_BYTES + INST_BYTES);
    float*    ptot = psq + PART_ELEMS;
    float*    pein = ptot + PART_ELEMS;

    (void)in_sizes; (void)n_in; (void)out_size; (void)ws_size;

    mic_prep<<<(BB * NN + 255) / 256, 256, 0, stream>>>(x, label, xh, inst);

    // waves = B * (N/16) * NCHUNK = 4096; 8 waves per 256-thread block
    const int nblocks = (BB * (NN / 16) * NCHUNK) / 8;   // 512
    mic_main<<<nblocks, 256, 0, stream>>>(xh, inst, psq, ptot, pein);

    mic_final<<<1, 128, 0, stream>>>(inst, psq, ptot, pein, out);
}
